// TransformerBlock_10797547782456
// MI455X (gfx1250) — matrix operand order repaired and checked by reading, not compiled, not run on hardware
//
#include <hip/hip_runtime.h>
#include <cstdint>
#include <cstddef>

// ---------------------------------------------------------------------------
// MI455X / gfx1250 transformer block forward.
//   All GEMMs + attention einsums use v_wmma_f32_16x16x32_bf16 (wave32).
//   Staging uses GLOBAL_LOAD_ASYNC_TO_LDS_B128 (ASYNCcnt) when available,
//   with a double-buffered LDS pipeline in the GEMM.
// ---------------------------------------------------------------------------

typedef __bf16 bf16;
typedef __attribute__((ext_vector_type(16))) __bf16 bf16x16;
typedef __attribute__((ext_vector_type(8)))  float  f32x8;

#if defined(__has_builtin)
#if __has_builtin(__builtin_amdgcn_global_load_async_to_lds_b128)
#define HAVE_ASYNC_LDS 1
#endif
#endif
#ifndef HAVE_ASYNC_LDS
#define HAVE_ASYNC_LDS 0
#endif

#if HAVE_ASYNC_LDS
// Builtin signature (from compiler diagnostic): first arg is
//   int __attribute__((vector_size(16))) addrspace(1)*  (global src),
// second is the LDS destination, then imm offset and imm cpol.
typedef int v4i_gcc __attribute__((vector_size(16)));
typedef __attribute__((address_space(1))) v4i_gcc* g_v4i_ptr;
typedef __attribute__((address_space(3))) v4i_gcc* l_v4i_ptr;
#if __has_builtin(__builtin_amdgcn_s_wait_asynccnt)
#define WAIT_ASYNC(n) __builtin_amdgcn_s_wait_asynccnt(n)
#else
#define WAIT_ASYNC(n) asm volatile("s_wait_asynccnt %0" ::"i"(n) : "memory")
#endif
#else
#define WAIT_ASYNC(n)
#endif

// Copy 16B global -> LDS. Async (no VGPR round-trip) when the builtin exists.
static __device__ __forceinline__ void cp16_g2l(bf16* l, const bf16* g) {
#if HAVE_ASYNC_LDS
  __builtin_amdgcn_global_load_async_to_lds_b128(
      (g_v4i_ptr)(uintptr_t)g, (l_v4i_ptr)(uintptr_t)l, 0, 0);
#else
  *reinterpret_cast<float4*>(l) = *reinterpret_cast<const float4*>(g);
#endif
}

// Load a 16-element bf16 fragment as two 16B chunks (ds_load_b128 x2).
static __device__ __forceinline__ bf16x16 ld_frag(const bf16* p0, const bf16* p1) {
  union { bf16x16 v; float4 q[2]; } u;
  u.q[0] = *reinterpret_cast<const float4*>(p0);
  u.q[1] = *reinterpret_cast<const float4*>(p1);
  return u.v;
}

static __device__ __forceinline__ f32x8 wmma_bf16(bf16x16 a, bf16x16 b, f32x8 c) {
  return __builtin_amdgcn_wmma_f32_16x16x32_bf16(false, a, false, b, (short)0, c,
                                                 false, false);
}

// ---------------------------------------------------------------------------
// Weight transpose + f32->bf16 convert:  W[K][N] (row major) -> Wt[N][K]
// ---------------------------------------------------------------------------
__global__ __launch_bounds__(256) void wt_transpose_kernel(
    const float* __restrict__ W, bf16* __restrict__ Wt, int K, int N) {
  __shared__ float t[32][33];
  const int n0 = blockIdx.x * 32;
  const int k0 = blockIdx.y * 32;
  const int tx = threadIdx.x;       // 0..31
  const int ty = threadIdx.y;       // 0..7
#pragma unroll
  for (int r = 0; r < 4; ++r)
    t[ty + r * 8][tx] = W[(size_t)(k0 + ty + r * 8) * N + n0 + tx];
  __syncthreads();
#pragma unroll
  for (int r = 0; r < 4; ++r)
    Wt[(size_t)(n0 + ty + r * 8) * K + k0 + tx] = (bf16)t[tx][ty + r * 8];
}

// ---------------------------------------------------------------------------
// LayerNorm over C=768, one block per row, f32 in -> bf16 out.
// ---------------------------------------------------------------------------
__global__ __launch_bounds__(256) void ln_kernel(
    const float* __restrict__ x, const float* __restrict__ g,
    const float* __restrict__ b, bf16* __restrict__ out) {
  const int row = blockIdx.x;
  const int tid = threadIdx.x;
  const float* xr = x + (size_t)row * 768;
  float v0 = xr[tid], v1 = xr[tid + 256], v2 = xr[tid + 512];

  __shared__ float red[256];
  red[tid] = v0 + v1 + v2;
  __syncthreads();
#pragma unroll
  for (int off = 128; off > 0; off >>= 1) {
    if (tid < off) red[tid] += red[tid + off];
    __syncthreads();
  }
  const float mu = red[0] * (1.0f / 768.0f);
  __syncthreads();

  const float d0 = v0 - mu, d1 = v1 - mu, d2 = v2 - mu;
  red[tid] = d0 * d0 + d1 * d1 + d2 * d2;
  __syncthreads();
#pragma unroll
  for (int off = 128; off > 0; off >>= 1) {
    if (tid < off) red[tid] += red[tid + off];
    __syncthreads();
  }
  const float rstd = rsqrtf(red[0] * (1.0f / 768.0f) + 1e-5f);

  bf16* orow = out + (size_t)row * 768;
  orow[tid]       = (bf16)(d0 * rstd * g[tid]       + b[tid]);
  orow[tid + 256] = (bf16)(d1 * rstd * g[tid + 256] + b[tid + 256]);
  orow[tid + 512] = (bf16)(d2 * rstd * g[tid + 512] + b[tid + 512]);
}

// ---------------------------------------------------------------------------
// WMMA GEMM:  out[M][N] = A[M][K] (bf16) @ Bt[N][K]^T (bf16) (+bias)(+resid)(GELU)
// 256 threads (8 waves); BM=128 x BN=128 x BK=32; double-buffered async staging.
// Wave (wm 0..3, wn 0..1) computes a 32x64 tile = 2x4 WMMA 16x16 tiles.
// ---------------------------------------------------------------------------
constexpr int LDA = 56;  // bf16 elements; 112B row -> 16B aligned, bank-spread

template <bool GELU, bool OUTBF16>
__global__ __launch_bounds__(256) void gemm_kernel(
    const bf16* __restrict__ A, const bf16* __restrict__ Bt,
    const float* __restrict__ bias, const float* __restrict__ resid,
    float* __restrict__ outF, bf16* __restrict__ outB,
    int M, int N, int K) {
  __shared__ bf16 sA[2][128 * LDA];
  __shared__ bf16 sB[2][128 * LDA];

  const int tid  = threadIdx.x;
  const int lane = tid & 31;
  const int m16  = lane & 15;
  const int half = lane >> 4;
  const int wave = __builtin_amdgcn_readfirstlane(tid >> 5);
  const int wm   = wave >> 1;   // 0..3
  const int wn   = wave & 1;    // 0..1
  const int bm   = blockIdx.y * 128;
  const int bn   = blockIdx.x * 128;

  // Staging indices: 4 x 16B copies per thread per tile (2 for A, 2 for B).
  const int srow0 = tid >> 2;              // 0..63
  const int srow1 = srow0 + 64;            // 64..127
  const int skc   = (tid & 3) * 8;         // 0,8,16,24

  f32x8 acc[2][4];
#pragma unroll
  for (int i = 0; i < 2; ++i)
#pragma unroll
    for (int j = 0; j < 4; ++j) acc[i][j] = (f32x8){0.f,0.f,0.f,0.f,0.f,0.f,0.f,0.f};

  const int nk = K >> 5;

  // Prologue: stage tile 0 into buffer 0.
  cp16_g2l(&sA[0][srow0 * LDA + skc], &A[(size_t)(bm + srow0) * K + skc]);
  cp16_g2l(&sA[0][srow1 * LDA + skc], &A[(size_t)(bm + srow1) * K + skc]);
  cp16_g2l(&sB[0][srow0 * LDA + skc], &Bt[(size_t)(bn + srow0) * K + skc]);
  cp16_g2l(&sB[0][srow1 * LDA + skc], &Bt[(size_t)(bn + srow1) * K + skc]);

  for (int t = 0; t < nk; ++t) {
    const int cur = t & 1;
    __syncthreads();  // all waves done reading buf[1-cur] from tile t-1
    if (t + 1 < nk) {
      const int k0 = (t + 1) * 32;
      cp16_g2l(&sA[1 - cur][srow0 * LDA + skc], &A[(size_t)(bm + srow0) * K + k0 + skc]);
      cp16_g2l(&sA[1 - cur][srow1 * LDA + skc], &A[(size_t)(bm + srow1) * K + k0 + skc]);
      cp16_g2l(&sB[1 - cur][srow0 * LDA + skc], &Bt[(size_t)(bn + srow0) * K + k0 + skc]);
      cp16_g2l(&sB[1 - cur][srow1 * LDA + skc], &Bt[(size_t)(bn + srow1) * K + k0 + skc]);
      WAIT_ASYNC(4);  // async loads complete in order: tile t's 4 are done
    } else {
      WAIT_ASYNC(0);
    }
    __syncthreads();  // tile t visible to every wave

    bf16x16 af[2], bfrag[4];
#pragma unroll
    for (int mt = 0; mt < 2; ++mt) {
      const bf16* base = &sA[cur][(wm * 32 + mt * 16 + m16) * LDA + half * 8];
      af[mt] = ld_frag(base, base + 16);
    }
#pragma unroll
    for (int nt = 0; nt < 4; ++nt) {
      const bf16* base = &sB[cur][(wn * 64 + nt * 16 + m16) * LDA + half * 8];
      bfrag[nt] = ld_frag(base, base + 16);
    }
#pragma unroll
    for (int mt = 0; mt < 2; ++mt)
#pragma unroll
      for (int nt = 0; nt < 4; ++nt)
        acc[mt][nt] = wmma_bf16(af[mt], bfrag[nt], acc[mt][nt]);
  }

  // Epilogue. C/D layout: vgpr i -> row half*8+i, col = lane&15.
#pragma unroll
  for (int mt = 0; mt < 2; ++mt)
#pragma unroll
    for (int nt = 0; nt < 4; ++nt) {
      const int gcol = bn + wn * 64 + nt * 16 + m16;
      const float bv = bias ? bias[gcol] : 0.0f;
#pragma unroll
      for (int i = 0; i < 8; ++i) {
        const int grow = bm + wm * 32 + mt * 16 + half * 8 + i;
        float v = acc[mt][nt][i] + bv;
        if (resid) v += resid[(size_t)grow * N + gcol];
        if (GELU)  v = 0.5f * v * (1.0f + erff(v * 0.70710678118654752f));
        if (OUTBF16) outB[(size_t)grow * N + gcol] = (bf16)v;
        else         outF[(size_t)grow * N + gcol] = v;
      }
    }
}

// ---------------------------------------------------------------------------
// Causal flash attention. Grid: (T/64, H, B), 128 threads = 4 waves.
// Wave w owns query rows q0+16w .. +15. KV tiles of 32 keys.
// ---------------------------------------------------------------------------
constexpr int LDK = 72;  // K tile row stride (bf16): 144B, 16B aligned
constexpr int LDV = 56;  // V^T tile row stride
constexpr int LDP = 48;  // per-wave P buffer row stride

__global__ __launch_bounds__(128) void attn_kernel(
    const bf16* __restrict__ q, const bf16* __restrict__ k,
    const bf16* __restrict__ v, bf16* __restrict__ o, int T) {
  __shared__ bf16 sK[32 * LDK];      // [key][d]
  __shared__ bf16 sV[64 * LDV];      // [d][key]  (transposed)
  __shared__ bf16 sP[4][16 * LDP];   // per-wave P = exp(S-m), [row][key]

  const int bb   = blockIdx.z;
  const int hh   = blockIdx.y;
  const int q0   = blockIdx.x * 64;
  const int lane = threadIdx.x & 31;
  const int m16  = lane & 15;
  const int half = lane >> 4;
  const int wave = __builtin_amdgcn_readfirstlane(threadIdx.x >> 5);
  const int q0w  = q0 + wave * 16;
  const size_t base = ((size_t)bb * T) * 768 + (size_t)hh * 64;

  // Q fragments (A-layout), rows q0w..q0w+15, d = 0..63 in two k-steps.
  bf16x16 aQ[2];
#pragma unroll
  for (int s = 0; s < 2; ++s) {
    union { bf16x16 vv; bf16 e[16]; } u;
#pragma unroll
    for (int j = 0; j < 16; ++j) {
      const int kk = s * 32 + ((j >> 3) << 4) + half * 8 + (j & 7);
      u.e[j] = q[base + (size_t)(q0w + m16) * 768 + kk];
    }
    aQ[s] = u.vv;
  }

  f32x8 O[4];
#pragma unroll
  for (int nt = 0; nt < 4; ++nt) O[nt] = (f32x8){0.f,0.f,0.f,0.f,0.f,0.f,0.f,0.f};
  float mrow[8], lrow[8];
#pragma unroll
  for (int i = 0; i < 8; ++i) { mrow[i] = -1e30f; lrow[i] = 0.0f; }

  const int ntiles = (q0 + 64) >> 5;            // block-wide kv tiles
  const int wtiles = ((q0w + 15) >> 5) + 1;     // tiles this wave needs (scalar)

  // Staging indices: 2 x (K async copy + V transpose) per thread per tile.
  const int srow = threadIdx.x >> 3;            // key 0..15 (+16 on 2nd pass)
  const int sdc  = (threadIdx.x & 7) * 8;       // d 0,8,..,56

  for (int t = 0; t < ntiles; ++t) {
    const int kv0 = t * 32;
    __syncthreads();
#pragma unroll
    for (int it = 0; it < 2; ++it) {
      const int row = srow + it * 16;           // key 0..31
      // K tile: async global->LDS, row-major [key][d].
      cp16_g2l(&sK[row * LDK + sdc], &k[base + (size_t)(kv0 + row) * 768 + sdc]);
      // V tile: VGPR round-trip + transposed scatter into LDS.
      float4 raw =
          *reinterpret_cast<const float4*>(&v[base + (size_t)(kv0 + row) * 768 + sdc]);
      const bf16* e = reinterpret_cast<const bf16*>(&raw);
#pragma unroll
      for (int d2 = 0; d2 < 8; ++d2) sV[(sdc + d2) * LDV + row] = e[d2];
      if (t + 1 < ntiles) {  // prefetch next tile's K/V rows into cache
        __builtin_prefetch(&k[base + (size_t)(kv0 + 32 + row) * 768 + sdc], 0, 1);
        __builtin_prefetch(&v[base + (size_t)(kv0 + 32 + row) * 768 + sdc], 0, 1);
      }
    }
    WAIT_ASYNC(0);
    __syncthreads();
    if (t >= wtiles) continue;  // scalar, wave-uniform skip (EXEC stays full)

    // S = Q @ K^T : 16 x 32, two n-tiles of keys.
    f32x8 s0 = (f32x8){0.f,0.f,0.f,0.f,0.f,0.f,0.f,0.f};
    f32x8 s1 = s0;
#pragma unroll
    for (int s = 0; s < 2; ++s) {
      const bf16* k0p = &sK[m16 * LDK + s * 32 + half * 8];
      const bf16* k1p = &sK[(16 + m16) * LDK + s * 32 + half * 8];
      s0 = wmma_bf16(aQ[s], ld_frag(k0p, k0p + 16), s0);
      s1 = wmma_bf16(aQ[s], ld_frag(k1p, k1p + 16), s1);
    }

    // Online softmax (scale 1/sqrt(64) = 0.125), causal mask, P -> LDS (bf16).
    float alpha[8];
#pragma unroll
    for (int i = 0; i < 8; ++i) {
      const int qg = q0w + half * 8 + i;
      float v0 = s0[i] * 0.125f;
      float v1 = s1[i] * 0.125f;
      if (kv0 + m16      > qg) v0 = -1e30f;
      if (kv0 + 16 + m16 > qg) v1 = -1e30f;
      float rm = fmaxf(v0, v1);
#pragma unroll
      for (int msk = 1; msk < 16; msk <<= 1) rm = fmaxf(rm, __shfl_xor(rm, msk, 32));
      const float mn = fmaxf(mrow[i], rm);
      const float a_ = __expf(mrow[i] - mn);
      const float p0 = __expf(v0 - mn);
      const float p1 = __expf(v1 - mn);
      float ps = p0 + p1;
#pragma unroll
      for (int msk = 1; msk < 16; msk <<= 1) ps += __shfl_xor(ps, msk, 32);
      lrow[i] = lrow[i] * a_ + ps;
      mrow[i] = mn;
      alpha[i] = a_;
      sP[wave][(half * 8 + i) * LDP + m16]      = (bf16)p0;
      sP[wave][(half * 8 + i) * LDP + 16 + m16] = (bf16)p1;
    }
#pragma unroll
    for (int nt = 0; nt < 4; ++nt)
#pragma unroll
      for (int i = 0; i < 8; ++i) O[nt][i] *= alpha[i];

    // O += P @ V  (P: 16x32 from LDS in A-layout, V^T gives B-layout reads)
    const bf16* pp = &sP[wave][m16 * LDP + half * 8];
    const bf16x16 aP = ld_frag(pp, pp + 16);
#pragma unroll
    for (int nt = 0; nt < 4; ++nt) {
      const bf16* vp = &sV[(nt * 16 + m16) * LDV + half * 8];
      O[nt] = wmma_bf16(aP, ld_frag(vp, vp + 16), O[nt]);
    }
  }

  // Normalize and store bf16 [B,T,C] at head columns.
#pragma unroll
  for (int nt = 0; nt < 4; ++nt)
#pragma unroll
    for (int i = 0; i < 8; ++i) {
      const float ov = O[nt][i] / lrow[i];
      o[base + (size_t)(q0w + half * 8 + i) * 768 + nt * 16 + m16] = (bf16)ov;
    }
}

// ---------------------------------------------------------------------------
// Host-side orchestration (all on `stream`, graph-capture safe).
// ---------------------------------------------------------------------------
extern "C" void kernel_launch(void* const* d_in, const int* in_sizes, int n_in,
                              void* d_out, int out_size, void* d_ws, size_t ws_size,
                              hipStream_t stream) {
  (void)n_in; (void)out_size; (void)ws_size;
  constexpr int C = 768, H = 12, B = 2, C4 = 3072;
  const int T = in_sizes[0] / (B * C);
  const int M = B * T;

  const float* x   = (const float*)d_in[0];
  const float* Wq  = (const float*)d_in[1];
  const float* Wk  = (const float*)d_in[2];
  const float* Wv  = (const float*)d_in[3];
  const float* Wo  = (const float*)d_in[4];
  const float* bo  = (const float*)d_in[5];
  const float* W1  = (const float*)d_in[6];
  const float* b1  = (const float*)d_in[7];
  const float* W2  = (const float*)d_in[8];
  const float* b2  = (const float*)d_in[9];
  const float* g1  = (const float*)d_in[10];
  const float* be1 = (const float*)d_in[11];
  const float* g2  = (const float*)d_in[12];
  const float* be2 = (const float*)d_in[13];
  float* out = (float*)d_out;

  char* ws = (char*)d_ws;
  size_t off = 0;
  auto alloc = [&](size_t bytes) -> char* {
    char* p = ws + off;
    off += (bytes + 255) & ~(size_t)255;
    return p;
  };

  const size_t szCC = (size_t)C * C * sizeof(bf16);
  const size_t szC4 = (size_t)C * C4 * sizeof(bf16);
  bf16* wqT = (bf16*)alloc(szCC);
  bf16* wkT = (bf16*)alloc(szCC);
  bf16* wvT = (bf16*)alloc(szCC);
  bf16* woT = (bf16*)alloc(szCC);
  bf16* w1T = (bf16*)alloc(szC4);
  bf16* w2T = (bf16*)alloc(szC4);

  const size_t MC2 = (size_t)M * C * sizeof(bf16);
  char* act = alloc(5 * MC2);           // xn1 | q | k | v | attn ; later reused by h
  bf16* xn1   = (bf16*)(act);
  bf16* qb    = (bf16*)(act + 1 * MC2);
  bf16* kb    = (bf16*)(act + 2 * MC2);
  bf16* vb    = (bf16*)(act + 3 * MC2);
  bf16* attnb = (bf16*)(act + 4 * MC2);
  bf16* hB    = (bf16*)(act);           // M x 3072 bf16 = 4*MC2, overlays dead bufs

  float* x1  = (float*)alloc((size_t)M * C * sizeof(float));
  bf16*  xn2 = (bf16*)alloc(MC2);

  const dim3 tb(32, 8);
  wt_transpose_kernel<<<dim3(C / 32, C / 32), tb, 0, stream>>>(Wq, wqT, C, C);
  wt_transpose_kernel<<<dim3(C / 32, C / 32), tb, 0, stream>>>(Wk, wkT, C, C);
  wt_transpose_kernel<<<dim3(C / 32, C / 32), tb, 0, stream>>>(Wv, wvT, C, C);
  wt_transpose_kernel<<<dim3(C / 32, C / 32), tb, 0, stream>>>(Wo, woT, C, C);
  wt_transpose_kernel<<<dim3(C4 / 32, C / 32), tb, 0, stream>>>(W1, w1T, C, C4);
  wt_transpose_kernel<<<dim3(C / 32, C4 / 32), tb, 0, stream>>>(W2, w2T, C4, C);

  // x -> LN1 -> bf16
  ln_kernel<<<M, 256, 0, stream>>>(x, g1, be1, xn1);

  // q/k/v = xn1 @ W{q,k,v}
  const dim3 gq(C / 128, M / 128);
  gemm_kernel<false, true><<<gq, 256, 0, stream>>>(xn1, wqT, nullptr, nullptr,
                                                   nullptr, qb, M, C, C);
  gemm_kernel<false, true><<<gq, 256, 0, stream>>>(xn1, wkT, nullptr, nullptr,
                                                   nullptr, kb, M, C, C);
  gemm_kernel<false, true><<<gq, 256, 0, stream>>>(xn1, wvT, nullptr, nullptr,
                                                   nullptr, vb, M, C, C);

  // Causal flash attention per (batch, head, 64-row query tile)
  attn_kernel<<<dim3(T / 64, H, B), 128, 0, stream>>>(qb, kb, vb, attnb, T);

  // x1 = x + attn @ Wo + bo
  gemm_kernel<false, false><<<gq, 256, 0, stream>>>(attnb, woT, bo, x, x1,
                                                    nullptr, M, C, C);

  // LN2, MLP
  ln_kernel<<<M, 256, 0, stream>>>(x1, g2, be2, xn2);
  gemm_kernel<true, true><<<dim3(C4 / 128, M / 128), 256, 0, stream>>>(
      xn2, w1T, b1, nullptr, nullptr, hB, M, C4, C);
  gemm_kernel<false, false><<<gq, 256, 0, stream>>>(hB, w2T, b2, x1, out,
                                                    nullptr, M, C, C4);
}
